// RNNs_59622736003524
// MI455X (gfx1250) — matrix operand order repaired and checked by reading, not compiled, not run on hardware
//
#include <hip/hip_runtime.h>
#include <cstdint>
#include <cstddef>

// ---------------------------------------------------------------------------
// Types for CDNA5 WMMA + TDM
// ---------------------------------------------------------------------------
typedef __attribute__((ext_vector_type(16))) __bf16       v16bf;
typedef __attribute__((ext_vector_type(8)))  float        v8f;
typedef __attribute__((ext_vector_type(4)))  unsigned int u32x4;
typedef __attribute__((ext_vector_type(8)))  int          i32x8;
typedef __attribute__((ext_vector_type(4)))  int          i32x4;

// ---------------------------------------------------------------------------
// Problem constants (match reference)
// ---------------------------------------------------------------------------
constexpr int Bn    = 512;
constexpr int Tn    = 288;
constexpr int Fn    = 8;
constexpr int HDn   = 256;
constexpr int PREDn = 12;

constexpr int BTn   = 16;          // batch rows per workgroup (one 16-row M tile)
constexpr int NBLK  = Bn / BTn;    // 32 workgroups, fully independent in batch
constexpr int NTHR  = 512;         // 16 waves: wave w owns N-tile w (16x16)
constexpr int LSTR  = 264;         // padded LDS row stride (bf16 elems) = 528B
                                   // 528B/4 = 132 dwords, 132 % 64 = 4 -> 4-bank
                                   // step per row: 16-row b128 reads conflict-free

// ---------------------------------------------------------------------------
// Workspace layout (bytes)
// ---------------------------------------------------------------------------
constexpr size_t OWC   = 0;                        // Wc  f32 [256][8]  (W_ih0 @ W_in)
constexpr size_t OBC   = OWC   + (size_t)256*8*4;  // bc  f32 [256]
constexpr size_t OBL1  = OBC   + 256*4;            // bL1 f32 [256]
constexpr size_t OWHH0 = OBL1  + 256*4;            // bf16 [256][256] row-major (j,k)
constexpr size_t OWIH1 = OWHH0 + (size_t)256*256*2;
constexpr size_t OWHH1 = OWIH1 + (size_t)256*256*2;
constexpr size_t OZ    = OWHH1 + (size_t)256*256*2;// bf16 [B][T][HD]
// total ~75.9 MB

constexpr size_t SMEM_BYTES = (size_t)(2*256*LSTR + 2*BTn*LSTR) * 2; // 287232 B

// ---------------------------------------------------------------------------
// Helpers
// ---------------------------------------------------------------------------
__device__ __forceinline__ unsigned short f2bf(float f) {
    union { float f; unsigned u; } v; v.f = f;
    unsigned r = v.u + 0x7FFFu + ((v.u >> 16) & 1u);   // round-to-nearest-even
    return (unsigned short)(r >> 16);
}
__device__ __forceinline__ float bf2f(unsigned short h) {
    union { unsigned u; float f; } v; v.u = ((unsigned)h) << 16; return v.f;
}
__device__ __forceinline__ float fast_tanh(float x) {
    // tanh(x) = (e^{2x}-1)/(e^{2x}+1); v_exp_f32 + v_rcp_f32
    float e = __expf(2.0f * x);
    return (e - 1.0f) * __builtin_amdgcn_rcpf(e + 1.0f);
}

// TDM: load a 256x256 bf16 matrix from global into LDS, padding each 512B row
// to a 528B stride (pad_interval = 128 dwords -> enc 6, pad_amount = 4 dwords
// -> enc 3).  2D tensor => descriptor groups 2/3 (and trailing group) zeroed.
__device__ __forceinline__ void tdm_load_w(unsigned lds_off, const void* gsrc) {
#if __has_builtin(__builtin_amdgcn_tensor_load_to_lds)
    unsigned long long ga = (unsigned long long)(uintptr_t)gsrc;
    u32x4 g0;
    g0[0] = 1u;                                   // count=1, user descriptor
    g0[1] = lds_off;                              // LDS byte address
    g0[2] = (unsigned)ga;                         // global_addr[31:0]
    g0[3] = (unsigned)(ga >> 32) | (2u << 30);    // global_addr[56:32] | type=2
    i32x8 g1;
    g1[0] = (1 << 16)       // data_size = 2 bytes
          | (1 << 20)       // pad_enable
          | (6 << 22)       // pad_interval: 128 dwords
          | (3 << 25);      // pad_amount:   4 dwords
    g1[1] = (int)(256u << 16);   // tensor_dim0[15:0] in bits [63:48]
    g1[2] = (int)(256u << 16);   // tensor_dim1[15:0] in bits [95:80]
    g1[3] = (int)(256u << 16);   // tile_dim0 in bits [127:112]
    g1[4] = 256;                 // tile_dim1
    g1[5] = 256;                 // tensor_dim0_stride (elems), low 32
    g1[6] = 0;                   // dim0_stride hi | dim1_stride low16 (65536&0xffff=0)
    g1[7] = 1;                   // dim1_stride bits [47:16] = 65536>>16
    i32x4 z4 = {0, 0, 0, 0};
    i32x8 z8 = {0, 0, 0, 0, 0, 0, 0, 0};
    __builtin_amdgcn_tensor_load_to_lds(g0, g1, z4, z4, z8, 0);
#else
    (void)lds_off; (void)gsrc;
#endif
}

// ---------------------------------------------------------------------------
// Kernel 1: fold input projection into layer0: Wc = W_ih0 @ W_in,
//           bc = W_ih0 @ b_in + b_ih0 + b_hh0, bL1 = b_ih1 + b_hh1.
// ---------------------------------------------------------------------------
__global__ void rnn_prep_combine(const float* __restrict__ W_in,
                                 const float* __restrict__ b_in,
                                 const float* __restrict__ W_ih,
                                 const float* __restrict__ b_ih,
                                 const float* __restrict__ W_hh,
                                 const float* __restrict__ b_hh,
                                 float* __restrict__ Wc,
                                 float* __restrict__ bc,
                                 float* __restrict__ bL1) {
    int j = threadIdx.x;                   // 0..255
    float acc[Fn];
#pragma unroll
    for (int f = 0; f < Fn; ++f) acc[f] = 0.0f;
    float accb = 0.0f;
    for (int k = 0; k < HDn; ++k) {
        float w = W_ih[j * HDn + k];       // W_ih[0][j][k]
        accb += w * b_in[k];
#pragma unroll
        for (int f = 0; f < Fn; ++f) acc[f] += w * W_in[k * Fn + f];
    }
#pragma unroll
    for (int f = 0; f < Fn; ++f) Wc[j * Fn + f] = acc[f];
    bc[j]  = accb + b_ih[j] + b_hh[j];
    bL1[j] = b_ih[HDn + j] + b_hh[HDn + j];
}

// ---------------------------------------------------------------------------
// Kernel 2: convert the three recurrent weight matrices to bf16 (row-major j,k)
// ---------------------------------------------------------------------------
__global__ void rnn_conv_bf16(const float* __restrict__ W_ih,
                              const float* __restrict__ W_hh,
                              unsigned short* __restrict__ whh0,
                              unsigned short* __restrict__ wih1,
                              unsigned short* __restrict__ whh1) {
    int idx = blockIdx.x * blockDim.x + threadIdx.x;     // < 3*65536
    int which = idx >> 16;
    int off   = idx & 65535;
    if (which == 0)      whh0[off] = f2bf(W_hh[off]);
    else if (which == 1) wih1[off] = f2bf(W_ih[65536 + off]);
    else                 whh1[off] = f2bf(W_hh[65536 + off]);
}

// ---------------------------------------------------------------------------
// Kernel 3: Z0[b,t,:] = x[b,t,:] @ Wc^T + bc   (bf16 out, 75 MB, ~3us @ HBM)
// ---------------------------------------------------------------------------
__global__ void rnn_z0(const float* __restrict__ x,
                       const float* __restrict__ Wc,
                       const float* __restrict__ bc,
                       unsigned short* __restrict__ z) {
    size_t bt = blockIdx.x;                // 0 .. B*T-1
    int j = threadIdx.x;                   // 0..255
    const float* xp = x + bt * Fn;
    float acc = bc[j];
#pragma unroll
    for (int f = 0; f < Fn; ++f) acc += xp[f] * Wc[j * Fn + f];
    z[bt * HDn + j] = f2bf(acc);
}

// ---------------------------------------------------------------------------
// Kernel 4: persistent recurrence. 32 workgroups x 16 batch rows, 16 waves,
// wave w owns N-tile w. Per step:
//   phase1: h0 = tanh(Z0[t] + h0 @ Whh0^T)                (8  WMMA / wave)
//   phase2: h1 = tanh(bL1 + h0 @ Wih1^T + h1 @ Whh1^T)    (16 WMMA / wave)
//   if t in last PRED steps of ys[:-1]: out = relu(h1) @ W_out + b_out
// ---------------------------------------------------------------------------
__global__ void __launch_bounds__(NTHR, 1)
rnn_main(const unsigned short* __restrict__ zbase,
         const unsigned short* __restrict__ gWhh0,
         const unsigned short* __restrict__ gWih1,
         const unsigned short* __restrict__ gWhh1,
         const float* __restrict__ bL1,
         const float* __restrict__ W_out,
         const float* __restrict__ b_out,
         float* __restrict__ out) {
    extern __shared__ unsigned short smem[];
    unsigned short* sWhh0 = smem;                         // [256][LSTR]
    unsigned short* sWih1 = sWhh0 + 256 * LSTR;           // [256][LSTR]
    unsigned short* sH0   = sWih1 + 256 * LSTR;           // [BT][LSTR]
    unsigned short* sH1   = sH0   + BTn * LSTR;           // [BT][LSTR]

    const int tid  = threadIdx.x;
    const int wave = tid >> 5;         // 0..15 -> N tile
    const int lane = tid & 31;
    const int half = lane >> 4;        // lane half selects K sub-block / M+8
    const int l16  = lane & 15;
    const int b0   = blockIdx.x * BTn;

    // h0 = h1 = 0
    for (int i = tid; i < BTn * LSTR; i += NTHR) { sH0[i] = 0; sH1[i] = 0; }

    // One-time TDM loads of the two LDS-resident weights (wave 0 issues;
    // TDM pad feature writes the 264-element row stride for us).
    if (wave == 0) {
        tdm_load_w((unsigned)(uintptr_t)sWhh0, gWhh0);
        tdm_load_w((unsigned)(uintptr_t)sWih1, gWih1);
#if __has_builtin(__builtin_amdgcn_s_wait_tensorcnt)
        __builtin_amdgcn_s_wait_tensorcnt((short)0);
#endif
    }
    __syncthreads();

    // Per-wave fragment base pointers.
    // A frag (16x32 bf16): lane = row M=l16; 16 contiguous k from h*16.
    const unsigned short* aRow0 = sH0 + l16 * LSTR + half * 16;
    const unsigned short* aRow1 = sH1 + l16 * LSTR + half * 16;
    // B frag (32x16 bf16): lane = col N=l16 (= W row j); 16 contiguous k.
    const unsigned short* bWhh0 = sWhh0 + (wave * 16 + l16) * LSTR + half * 16;
    const unsigned short* bWih1 = sWih1 + (wave * 16 + l16) * LSTR + half * 16;
    const unsigned short* bWhh1 = gWhh1 + (size_t)(wave * 16 + l16) * HDn + half * 16;

    const float biasL1 = bL1[wave * 16 + l16];
    const float bout0  = b_out[0];

    union Frag { v16bf v; uint4 q[2]; };

    for (int t = 0; t < Tn - 1; ++t) {
        // ---- phase 1: layer 0 ------------------------------------------
        v8f acc0;
#pragma unroll
        for (int r = 0; r < 8; ++r) {
            int row = r + 8 * half;                     // C/D layout: M = r (+8)
            acc0[r] = bf2f(zbase[((size_t)(b0 + row) * Tn + t) * HDn
                                 + wave * 16 + l16]);
        }
        if (t + 1 < Tn - 1) {
            __builtin_prefetch(&zbase[((size_t)(b0 + 8 * half) * Tn + (t + 1)) * HDn
                                      + wave * 16], 0, 1);
        }
#pragma unroll
        for (int kc = 0; kc < 8; ++kc) {
            Frag a, b;
            a.q[0] = *(const uint4*)(aRow0 + kc * 32);
            a.q[1] = *(const uint4*)(aRow0 + kc * 32 + 8);
            b.q[0] = *(const uint4*)(bWhh0 + kc * 32);
            b.q[1] = *(const uint4*)(bWhh0 + kc * 32 + 8);
            acc0 = __builtin_amdgcn_wmma_f32_16x16x32_bf16(
                false, a.v, false, b.v, (short)0, acc0, false, false);
        }
        __syncthreads();                                // all h0 reads done
#pragma unroll
        for (int r = 0; r < 8; ++r)
            sH0[(r + 8 * half) * LSTR + wave * 16 + l16] = f2bf(fast_tanh(acc0[r]));
        __syncthreads();                                // h0_new visible

        // ---- phase 2: layer 1 (input + hidden GEMMs fused) --------------
        v8f acc1;
#pragma unroll
        for (int r = 0; r < 8; ++r) acc1[r] = biasL1;
#pragma unroll
        for (int kc = 0; kc < 8; ++kc) {
            Frag a, b;
            a.q[0] = *(const uint4*)(aRow0 + kc * 32);          // h0_new
            a.q[1] = *(const uint4*)(aRow0 + kc * 32 + 8);
            b.q[0] = *(const uint4*)(bWih1 + kc * 32);
            b.q[1] = *(const uint4*)(bWih1 + kc * 32 + 8);
            acc1 = __builtin_amdgcn_wmma_f32_16x16x32_bf16(
                false, a.v, false, b.v, (short)0, acc1, false, false);
            a.q[0] = *(const uint4*)(aRow1 + kc * 32);          // h1 (prev)
            a.q[1] = *(const uint4*)(aRow1 + kc * 32 + 8);
            b.q[0] = *(const uint4*)(bWhh1 + kc * 32);          // L2-streamed
            b.q[1] = *(const uint4*)(bWhh1 + kc * 32 + 8);
            acc1 = __builtin_amdgcn_wmma_f32_16x16x32_bf16(
                false, a.v, false, b.v, (short)0, acc1, false, false);
        }
        __syncthreads();                                // all h1 reads done
#pragma unroll
        for (int r = 0; r < 8; ++r)
            sH1[(r + 8 * half) * LSTR + wave * 16 + l16] = f2bf(fast_tanh(acc1[r]));
        __syncthreads();                                // h1_new visible

        // ---- output head on the last PRED kept steps --------------------
        if (t >= Tn - 1 - PREDn) {                      // t = 275 .. 286
            int row = wave;                             // one wave per batch row
            const unsigned short* hrow = sH1 + row * LSTR + lane * 8;
            float p = 0.0f;
#pragma unroll
            for (int i = 0; i < 8; ++i) {
                float v = bf2f(hrow[i]);
                v = v > 0.0f ? v : 0.0f;                // relu
                p += v * W_out[lane * 8 + i];
            }
#pragma unroll
            for (int off = 16; off >= 1; off >>= 1)
                p += __shfl_xor(p, off, 32);            // wave32 reduce
            if (lane == 0)
                out[(size_t)(b0 + row) * PREDn + (t - (Tn - 1 - PREDn))] = p + bout0;
        }
    }
}

// ---------------------------------------------------------------------------
// Host-side launcher
// ---------------------------------------------------------------------------
extern "C" void kernel_launch(void* const* d_in, const int* in_sizes, int n_in,
                              void* d_out, int out_size, void* d_ws, size_t ws_size,
                              hipStream_t stream) {
    (void)in_sizes; (void)n_in; (void)out_size; (void)ws_size;

    const float* x     = (const float*)d_in[0];   // [512,288,8]
    const float* W_in  = (const float*)d_in[1];   // [256,8]
    const float* b_in  = (const float*)d_in[2];   // [256]
    const float* W_ih  = (const float*)d_in[3];   // [2,256,256]
    const float* b_ih  = (const float*)d_in[4];   // [2,256]
    const float* W_hh  = (const float*)d_in[5];   // [2,256,256]
    const float* b_hh  = (const float*)d_in[6];   // [2,256]
    const float* W_out = (const float*)d_in[7];   // [1,256]
    const float* b_out = (const float*)d_in[8];   // [1]
    float* out = (float*)d_out;                   // [512,12,1]

    char* ws = (char*)d_ws;
    float*          Wc   = (float*)(ws + OWC);
    float*          bc   = (float*)(ws + OBC);
    float*          bL1  = (float*)(ws + OBL1);
    unsigned short* whh0 = (unsigned short*)(ws + OWHH0);
    unsigned short* wih1 = (unsigned short*)(ws + OWIH1);
    unsigned short* whh1 = (unsigned short*)(ws + OWHH1);
    unsigned short* z0   = (unsigned short*)(ws + OZ);

    rnn_prep_combine<<<1, 256, 0, stream>>>(W_in, b_in, W_ih, b_ih, W_hh, b_hh,
                                            Wc, bc, bL1);
    rnn_conv_bf16<<<(3 * 65536) / 256, 256, 0, stream>>>(W_ih, W_hh,
                                                         whh0, wih1, whh1);
    rnn_z0<<<Bn * Tn, 256, 0, stream>>>(x, Wc, bc, z0);

    (void)hipFuncSetAttribute(reinterpret_cast<const void*>(rnn_main),
                              hipFuncAttributeMaxDynamicSharedMemorySize,
                              (int)SMEM_BYTES);
    rnn_main<<<NBLK, NTHR, SMEM_BYTES, stream>>>(z0, whh0, wih1, whh1,
                                                 bL1, W_out, b_out, out);
}
